// EndToEndLoss_76536317215070
// MI455X (gfx1250) — matrix operand order repaired and checked by reading, not compiled, not run on hardware
//
#include <hip/hip_runtime.h>
#include <math.h>

#define N_SPK 1024
#define N_UTT 20
#define EMB   256
#define NROWS (N_SPK * N_UTT)     // 20480
#define ROWTILES (NROWS / 16)     // 1280

typedef __attribute__((ext_vector_type(16))) __bf16 bf16x16;
typedef __attribute__((ext_vector_type(8)))  __bf16 bf16x8;
typedef __attribute__((ext_vector_type(8)))  float  f32x8;

__device__ __forceinline__ void split_bf16(float x, __bf16& hi, __bf16& lo) {
  hi = (__bf16)x;                 // RTN to bf16
  lo = (__bf16)(x - (float)hi);   // residual captures next ~8 mantissa bits
}

// ---------------------------------------------------------------------------
// Kernel 1: per-speaker centroids, normalization, leave-one-out self terms.
// One block per speaker, one thread per embedding dim.
// ---------------------------------------------------------------------------
__global__ void __launch_bounds__(256)
ge2e_prep_kernel(const float* __restrict__ inp,
                 const float* __restrict__ wp, const float* __restrict__ bp,
                 __bf16* __restrict__ Uhi, __bf16* __restrict__ Ulo,
                 __bf16* __restrict__ Chi, __bf16* __restrict__ Clo,
                 float* __restrict__ self_sim, float* __restrict__ own_sim) {
  __shared__ float s1[256];
  __shared__ float s2[256];
  const int n = blockIdx.x;
  const int d = threadIdx.x;
  const float w = wp[0], b = bp[0];

  // load all 20 utterances of this speaker for dim d
  float u[N_UTT];
  const float* base = inp + (size_t)n * N_UTT * EMB + d;
  #pragma unroll
  for (int m = 0; m < N_UTT; ++m) u[m] = base[(size_t)m * EMB];

  // centroid (mean over utterances)
  float c = 0.f;
  #pragma unroll
  for (int m = 0; m < N_UTT; ++m) c += u[m];
  c *= (1.0f / (float)N_UTT);

  // reduce ||c||^2
  s1[d] = c * c;
  __syncthreads();
  for (int off = 128; off > 0; off >>= 1) {
    if (d < off) s1[d] += s1[d + off];
    __syncthreads();
  }
  const float cc = s1[0];
  __syncthreads();
  const float inv_cn = 1.0f / sqrtf(cc);

  // normalized centroid, bf16 hi/lo
  {
    __bf16 h, l;
    split_bf16(c * inv_cn, h, l);
    Chi[(size_t)n * EMB + d] = h;
    Clo[(size_t)n * EMB + d] = l;
  }

  for (int m = 0; m < N_UTT; ++m) {
    // reduce ||u||^2 and u.c simultaneously
    s1[d] = u[m] * u[m];
    s2[d] = u[m] * c;
    __syncthreads();
    for (int off = 128; off > 0; off >>= 1) {
      if (d < off) { s1[d] += s1[d + off]; s2[d] += s2[d + off]; }
      __syncthreads();
    }
    const float uu = s1[0];
    const float uc = s2[0];
    __syncthreads();

    const float inv_un = 1.0f / sqrtf(uu);
    __bf16 h, l;
    split_bf16(u[m] * inv_un, h, l);
    const size_t r = (size_t)(n * N_UTT + m);
    Uhi[r * EMB + d] = h;
    Ulo[r * EMB + d] = l;

    if (d == 0) {
      // excl = (c*M - u)/(M-1); closed-form cosines (all fp32):
      const float M = (float)N_UTT;
      const float num = M * uc - uu;                               // (M-1)*(u.excl)
      const float den = sqrtf(M * M * cc - 2.0f * M * uc + uu);    // (M-1)*||excl||
      const float cs_self = num * inv_un / den;
      const float cos_own = uc * inv_un * inv_cn;
      self_sim[r] = w * cs_self + b;
      own_sim[r]  = w * cos_own + b;
    }
  }
}

// ---------------------------------------------------------------------------
// Kernel 2: fused similarity GEMM (split-bf16 WMMA) + streaming exp-sum +
// own-speaker correction + per-tile loss partials. One wave32 per 16 rows.
// ---------------------------------------------------------------------------
__global__ void __launch_bounds__(32)
ge2e_gemm_lse_kernel(const __bf16* __restrict__ Uhi, const __bf16* __restrict__ Ulo,
                     const __bf16* __restrict__ Chi, const __bf16* __restrict__ Clo,
                     const float* __restrict__ self_sim, const float* __restrict__ own_sim,
                     const float* __restrict__ wp, const float* __restrict__ bp,
                     float* __restrict__ partial) {
  const int rt   = blockIdx.x;
  const int lane = threadIdx.x;
  const int lm   = lane & 15;
  const int half = lane >> 4;
  const float w = wp[0], b = bp[0];

  // A fragments (16x32 bf16): lanes 0-15 hold K{0..15},
  // lanes 16-31 hold K{16..31} of the chunk. Two 16B loads each.
  bf16x16 Ah[8], Al[8];
  {
    const int row = rt * 16 + lm;
    const __bf16* phi = Uhi + (size_t)row * EMB;
    const __bf16* plo = Ulo + (size_t)row * EMB;
    #pragma unroll
    for (int kc = 0; kc < 8; ++kc) {
      const int d0 = kc * 32 + 16 * half;
      bf16x8 a0 = *(const bf16x8*)(phi + d0);
      bf16x8 a1 = *(const bf16x8*)(phi + d0 + 8);
      bf16x8 l0 = *(const bf16x8*)(plo + d0);
      bf16x8 l1 = *(const bf16x8*)(plo + d0 + 8);
      #pragma unroll
      for (int i = 0; i < 8; ++i) {
        Ah[kc][i] = a0[i]; Ah[kc][i + 8] = a1[i];
        Al[kc][i] = l0[i]; Al[kc][i + 8] = l1[i];
      }
    }
  }

  float accExp[8];
  #pragma unroll
  for (int i = 0; i < 8; ++i) accExp[i] = 0.f;

  // sim = w*cos + b <= w + b = 5  -> exp never overflows: no running max needed.
  #pragma unroll 1
  for (int ct = 0; ct < 64; ++ct) {
    const int col = ct * 16 + lm;
    // B fragment (32x16 bf16): lane holds contiguous K{0..15} (half 0) or
    // K{16..31} (half 1) of its column -> single 32B load.
    const __bf16* qhi = Chi + (size_t)col * EMB + 16 * half;
    const __bf16* qlo = Clo + (size_t)col * EMB + 16 * half;
    f32x8 C = {0.f, 0.f, 0.f, 0.f, 0.f, 0.f, 0.f, 0.f};
    #pragma unroll
    for (int kc = 0; kc < 8; ++kc) {
      bf16x16 Bh = *(const bf16x16*)(qhi + kc * 32);
      bf16x16 Bl = *(const bf16x16*)(qlo + kc * 32);
      // split product: ah*bh + ah*bl + al*bh  (~fp32 accuracy)
      C = __builtin_amdgcn_wmma_f32_16x16x32_bf16(false, Ah[kc], false, Bh, (short)0, C, false, false);
      C = __builtin_amdgcn_wmma_f32_16x16x32_bf16(false, Ah[kc], false, Bl, (short)0, C, false, false);
      C = __builtin_amdgcn_wmma_f32_16x16x32_bf16(false, Al[kc], false, Bh, (short)0, C, false, false);
    }
    // C element i on this lane = (row i + 8*half, col lm) of the tile
    #pragma unroll
    for (int i = 0; i < 8; ++i)
      accExp[i] += __expf(w * C[i] + b);
  }

  // reduce exp-sums across the 16 lanes of each half (columns of the tile)
  #pragma unroll
  for (int off = 1; off < 16; off <<= 1) {
    #pragma unroll
    for (int i = 0; i < 8; ++i)
      accExp[i] += __shfl_xor(accExp[i], off, 32);
  }

  // all lanes in a half now hold identical row sums; compute loss contribs
  float local = 0.f;
  #pragma unroll
  for (int i = 0; i < 8; ++i) {
    const int r = rt * 16 + 8 * half + i;
    const float ss = self_sim[r];
    const float os = own_sim[r];
    const float S  = accExp[i] - __expf(os) + __expf(ss);  // swap own col -> LOO self
    local += logf(S) - ss;                                  // -logp[target]
  }
  const float tot = local + __shfl_xor(local, 16, 32);
  if (lane == 0) partial[rt] = tot;
}

// ---------------------------------------------------------------------------
// Kernel 3: deterministic final reduction of 1280 partials.
// ---------------------------------------------------------------------------
__global__ void __launch_bounds__(256)
ge2e_reduce_kernel(const float* __restrict__ partial, float* __restrict__ out) {
  __shared__ float s[256];
  const int t = threadIdx.x;
  float v = 0.f;
  for (int i = t; i < ROWTILES; i += 256) v += partial[i];
  s[t] = v;
  __syncthreads();
  for (int off = 128; off > 0; off >>= 1) {
    if (t < off) s[t] += s[t + off];
    __syncthreads();
  }
  if (t == 0) out[0] = s[0];
}

// ---------------------------------------------------------------------------
extern "C" void kernel_launch(void* const* d_in, const int* in_sizes, int n_in,
                              void* d_out, int out_size, void* d_ws, size_t ws_size,
                              hipStream_t stream) {
  (void)in_sizes; (void)n_in; (void)out_size; (void)ws_size;
  const float* inp = (const float*)d_in[0];
  const float* wp  = (const float*)d_in[1];
  const float* bp  = (const float*)d_in[2];
  float* out = (float*)d_out;

  char* ws = (char*)d_ws;
  size_t off = 0;
  auto carve = [&](size_t bytes) -> void* {
    off = (off + 255) & ~(size_t)255;
    void* p = ws + off;
    off += bytes;
    return p;
  };
  __bf16* Uhi = (__bf16*)carve((size_t)NROWS * EMB * sizeof(__bf16));
  __bf16* Ulo = (__bf16*)carve((size_t)NROWS * EMB * sizeof(__bf16));
  __bf16* Chi = (__bf16*)carve((size_t)N_SPK * EMB * sizeof(__bf16));
  __bf16* Clo = (__bf16*)carve((size_t)N_SPK * EMB * sizeof(__bf16));
  float* self_sim = (float*)carve((size_t)NROWS * sizeof(float));
  float* own_sim  = (float*)carve((size_t)NROWS * sizeof(float));
  float* partial  = (float*)carve((size_t)ROWTILES * sizeof(float));

  ge2e_prep_kernel<<<N_SPK, 256, 0, stream>>>(inp, wp, bp, Uhi, Ulo, Chi, Clo,
                                              self_sim, own_sim);
  ge2e_gemm_lse_kernel<<<ROWTILES, 32, 0, stream>>>(Uhi, Ulo, Chi, Clo,
                                                    self_sim, own_sim, wp, bp,
                                                    partial);
  ge2e_reduce_kernel<<<1, 256, 0, stream>>>(partial, out);
}
